// CrossVariableMixingConv_11149735100514
// MI455X (gfx1250) — compile-verified, matrix operand order checked by reading, not run on hardware
//
#include <hip/hip_runtime.h>
#include <hip/hip_bf16.h>

// CDNA5 WMMA vector types (wave32)
typedef __attribute__((ext_vector_type(16))) _Float16 v16h;
typedef __attribute__((ext_vector_type(8)))  float    v8f;

#define B_DIM   16
#define L_DIM   720
#define C_DIM   862
#define RANK    8
#define KTAPS   7
#define NROWS   (B_DIM * L_DIM)     // 11520
#define TPB     256
#define NWAVES  (TPB / 32)          // 8
#define TILE    32                  // positions per wave-iteration (1 wmma)
#define NTILES  ((C_DIM + TILE - 1) / TILE)   // 27
#define XS_LEN  (3 + C_DIM + 40)    // front halo 3, generous zero tail
#define EPS_LN  1e-5f

// Branch-free erf-based GELU (Abramowitz-Stegun 7.1.26, |err| <= 1.5e-7).
__device__ __forceinline__ float gelu_erf_fast(float v) {
    const float t = v * 0.70710678118654752440f;   // v / sqrt(2)
    const float a = __builtin_fabsf(t);
    const float k = __builtin_amdgcn_rcpf(__builtin_fmaf(0.3275911f, a, 1.0f));
    float p = __builtin_fmaf(1.061405429f, k, -1.453152027f);
    p = __builtin_fmaf(p, k,  1.421413741f);
    p = __builtin_fmaf(p, k, -0.284496736f);
    p = __builtin_fmaf(p, k,  0.254829592f);
    p = p * k;
    const float e     = __builtin_amdgcn_exp2f((a * a) * -1.4426950408889634f);
    const float erf_a = __builtin_fmaf(-p, e, 1.0f);
    const float erf_t = __builtin_copysignf(erf_a, t);
    return 0.5f * v * (1.0f + erf_t);
}

// Async global -> LDS copy of one dword per lane (CDNA5 GLOBAL_LOAD_ASYNC_TO_LDS,
// GVS mode: LDS byte-address VGPR, 32-bit byte-offset VGPR, SGPR64 base).
__device__ __forceinline__ void async_ld_b32(unsigned lds_byte_addr,
                                             const float* __restrict__ sbase,
                                             int byte_off) {
    asm volatile("global_load_async_to_lds_b32 %0, %1, %2"
                 :
                 : "v"(lds_byte_addr), "v"(byte_off), "s"(sbase)
                 : "memory");
}

__device__ __forceinline__ void wait_async0() {
    asm volatile("s_wait_asynccnt 0x0" ::: "memory");
}

__global__ __launch_bounds__(TPB)
void cross_var_mix_kernel(const float* __restrict__ x,      // (B,L,C)
                          const float* __restrict__ Wup,    // (RANK,1,K)
                          const float* __restrict__ bup,    // (RANK)
                          const float* __restrict__ Wdown,  // (1,RANK,1)
                          const float* __restrict__ bdown,  // (1)
                          const float* __restrict__ gamma,  // (C)
                          const float* __restrict__ beta,   // (C)
                          float* __restrict__ out)          // (B,L,C)
{
    __shared__ float xs[XS_LEN];     // x row with zero halo; xs[3+i] = x[i]
    __shared__ float ys[C_DIM];      // pre-norm y
    __shared__ float red_s[NWAVES], red_s2[NWAVES];

    const int row  = blockIdx.x;
    const int tid  = threadIdx.x;
    const int lane = tid & 31;
    const int wave = tid >> 5;

    const float* __restrict__ xrow = x + (size_t)row * C_DIM;

    // ---- Zero halos (non-overlapping with async payload region) ----------
    if (tid < 3) xs[tid] = 0.0f;
    for (int i = 3 + C_DIM + tid; i < XS_LEN; i += TPB) xs[i] = 0.0f;

    // ---- Async-stage row into LDS (ASYNCcnt-tracked, no VGPR bounce) -----
    for (int i = tid; i < C_DIM; i += TPB)
        async_ld_b32((unsigned)(uintptr_t)&xs[3 + i], xrow, i * 4);

    // ---- Build block-diagonal A (weights) once ---------------------------
    // 16-bit A (16x32): lane l<16 -> row M=l holding K=0..7 in elems 0..7;
    // lane l>=16 -> row M=l-16 holding K=8..15 in elems 0..7.
    // Rows 0..7 carry Wup vs K=0..6  (tile-0 taps): lanes 0..7.
    // Rows 8..15 carry Wup vs K=8..14 (tile-1 taps): lanes 24..31.
    // Everything else zero, so unused B slots (K=7,15,>=16) are annihilated.
    v16h amat;
#pragma unroll
    for (int e = 0; e < 16; ++e) amat[e] = (_Float16)0.0f;
    {
        const int r = (lane < 8) ? lane : ((lane >= 24) ? (lane - 24) : -1);
        if (r >= 0) {
#pragma unroll
            for (int k = 0; k < KTAPS; ++k)
                amat[k] = (_Float16)Wup[r * KTAPS + k];
        }
    }

    float bupv[RANK], wdnv[RANK];
#pragma unroll
    for (int r = 0; r < RANK; ++r) { bupv[r] = bup[r]; wdnv[r] = Wdown[r]; }
    const float bdnv = bdown[0];

    // B high elements (K>=16 region) stay zero for the whole kernel.
    v16h bmat;
#pragma unroll
    for (int e = 0; e < 16; ++e) bmat[e] = (_Float16)0.0f;

    wait_async0();
    __syncthreads();

    // ---- Conv(1->8,k=7) + GELU + 8->1 down + residual, 32 pos / wmma ----
    for (int t = wave; t < NTILES; t += NWAVES) {
        const int base = t * TILE;          // position of lane 0
        // v[k] = x[base+lane + k - 3]; this IS lane's B column:
        //   lanes 0-15  -> tile-0 column (K=0..6),
        //   lanes 16-31 -> tile-1 column (K=8..14). No shuffles needed.
        float v[KTAPS];
#pragma unroll
        for (int k = 0; k < KTAPS; ++k)
            v[k] = xs[base + lane + k];     // xs has +3 halo baked in

#pragma unroll
        for (int k = 0; k < KTAPS; ++k)
            bmat[k] = (_Float16)v[k];       // elems 7..15 remain zero

        // One WMMA: D rows 0-7 = tile-0 convs, rows 8-15 = tile-1 convs.
        // C/D layout => lane l holds h[rank j][base+l] in d[j] for ALL lanes.
        v8f d = {};
        d = __builtin_amdgcn_wmma_f32_16x16x32_f16(false, amat, false, bmat,
                                                   (short)0, d, false, false);

        float acc = 0.0f;
#pragma unroll
        for (int j = 0; j < RANK; ++j)
            acc += gelu_erf_fast(d[j] + bupv[j]) * wdnv[j];

        const int pos = base + lane;
        const float yv = v[3] + acc + bdnv;  // v[3] == x[pos] (residual)
        if (pos < C_DIM) ys[pos] = yv;
    }
    __syncthreads();

    // ---- LayerNorm over C -----------------------------------------------
    float s = 0.0f, s2 = 0.0f;
    for (int i = tid; i < C_DIM; i += TPB) {
        const float yv = ys[i];
        s += yv; s2 += yv * yv;
    }
#pragma unroll
    for (int off = 16; off > 0; off >>= 1) {
        s  += __shfl_down(s,  off, 32);
        s2 += __shfl_down(s2, off, 32);
    }
    if (lane == 0) { red_s[wave] = s; red_s2[wave] = s2; }
    __syncthreads();

    float ts = 0.0f, ts2 = 0.0f;
#pragma unroll
    for (int w = 0; w < NWAVES; ++w) { ts += red_s[w]; ts2 += red_s2[w]; }
    const float invC = 1.0f / (float)C_DIM;
    const float mu   = ts * invC;
    const float var  = __builtin_fmaxf(ts2 * invC - mu * mu, 0.0f);
    const float rsig = __builtin_amdgcn_rsqf(var + EPS_LN);

    float* __restrict__ orow = out + (size_t)row * C_DIM;
    for (int i = tid; i < C_DIM; i += TPB)
        orow[i] = (ys[i] - mu) * rsig * gamma[i] + beta[i];
}

extern "C" void kernel_launch(void* const* d_in, const int* in_sizes, int n_in,
                              void* d_out, int out_size, void* d_ws, size_t ws_size,
                              hipStream_t stream) {
    (void)in_sizes; (void)n_in; (void)out_size; (void)d_ws; (void)ws_size;
    const float* x     = (const float*)d_in[0];
    const float* Wup   = (const float*)d_in[1];
    const float* bup   = (const float*)d_in[2];
    const float* Wdown = (const float*)d_in[3];
    const float* bdown = (const float*)d_in[4];
    const float* gamma = (const float*)d_in[5];
    const float* beta  = (const float*)d_in[6];
    float* out = (float*)d_out;

    cross_var_mix_kernel<<<NROWS, TPB, 0, stream>>>(
        x, Wup, bup, Wdown, bdown, gamma, beta, out);
}
